// ModelSimplest_78005196030328
// MI455X (gfx1250) — compile-verified, matrix operand order checked by reading, not run on hardware
//
#include <hip/hip_runtime.h>
#include <hip/hip_bf16.h>
#include <math.h>

// ---------------------------------------------------------------------------
// 4D conv(1->3, 13^4) + bias + ReLU + Linear(3888->1) + sigmoid, B=512
// Implicit GEMM on v_wmma_f32_16x16x32_bf16 (gfx1250 / CDNA5, wave32).
//
//   M = 16 output positions per wave-tile (81 tiles of 6^4=1296 positions)
//   N = 16 (channels 0..2 real, rest zero-padded in weight fragments)
//   K = 13^4 re-indexed as 2197 (ix,iy,iz)-cells x 16 padded iw -> 1099 steps
//
// x[b] staged per block into LDS as bf16 [18][18][18][21] (244,944 B).
// Weight B-fragments pre-packed in WMMA lane layout (workspace, L2-resident).
// Per-step (cell0,cell1) LDS offsets come from a precomputed uniform table
// (kills the scalar carry chain seen in round 1).
// ---------------------------------------------------------------------------

typedef __attribute__((ext_vector_type(16))) __bf16 bf16x16;
typedef __attribute__((ext_vector_type(8)))  float  floatx8;

#define KCELLS   2197      // 13^3 (ix,iy,iz) cells
#define KSTEPS   1099      // ceil(2197/2): one wmma (K=32) eats 2 cells of 16 k'
#define SXPAD    122472    // 18*18*18*21 bf16 elements in LDS
#define XIN      104976    // 18^4
#define KW       28561     // 13^4
#define NTILES   81        // 1296 / 16
#define NFRAG    (KSTEPS * 32)          // B-fragment count (one per step,lane)

__device__ __forceinline__ int cell_off(int cell) {
    // cell -> ((ix*18+iy)*18+iz)*21 element offset in the padded LDS image
    int ix = cell / 169;
    int r  = cell - ix * 169;
    int iy = r / 13;
    int iz = r - iy * 13;
    return ((ix * 18 + iy) * 18 + iz) * 21;
}

// --- kernel 1: pack weights into WMMA B-fragment layout + step-offset table --
__global__ __launch_bounds__(256, 1)
void conv4d_prep(const float* __restrict__ w, __bf16* __restrict__ wsB,
                 int2* __restrict__ ctab) {
    int id = blockIdx.x * 256 + threadIdx.x;

    if (id < NFRAG) {
        int t    = id >> 5;
        int lane = id & 31;
        int n    = lane & 15;
        int iwb  = (lane >= 16) ? 8 : 0;
        bf16x16 pack;
#pragma unroll
        for (int j = 0; j < 16; ++j) {
            int grp  = j >> 3;                 // 0 -> cell 2t, 1 -> cell 2t+1
            int iw   = iwb + (j & 7);          // padded inner K index 0..15
            int cell = 2 * t + grp;
            float v = 0.f;
            if (n < 3 && iw < 13 && cell < KCELLS)
                v = w[n * KW + cell * 13 + iw];  // k = ((ix*13+iy)*13+iz)*13+iw
            pack[j] = (__bf16)v;
        }
        reinterpret_cast<bf16x16*>(wsB)[id] = pack;   // 32B store
    } else if (id < NFRAG + KSTEPS) {
        int t  = id - NFRAG;
        int c0 = 2 * t;
        int c1 = 2 * t + 1;
        if (c1 >= KCELLS) c1 = KCELLS - 1;    // B frag there is 0 anyway
        ctab[t] = make_int2(cell_off(c0), cell_off(c1));
    }
}

// --- kernel 2: one block per batch element ---------------------------------
__global__ __launch_bounds__(256, 1)
void conv4d_main(const float*  __restrict__ x,
                 const __bf16* __restrict__ wsB,
                 const int2*   __restrict__ ctab,
                 const float*  __restrict__ conv_b,
                 const float*  __restrict__ lin_w,
                 const float*  __restrict__ lin_b,
                 float*        __restrict__ out) {
    __shared__ __bf16 sxs[SXPAD];   // x[b] as bf16, innermost dim padded 18->21
    __shared__ float  ssum;

    const int b    = blockIdx.x;
    const int tid  = threadIdx.x;
    const int wave = tid >> 5;
    const int lane = tid & 31;

    // ---- stage x[b] into LDS (fp32 -> bf16), zero the 3-wide W pad ----
    const float* xb = x + (size_t)b * XIN;
    for (int i = tid; i < SXPAD; i += 256) {
        unsigned rest = (unsigned)i / 21u;
        unsigned wp   = (unsigned)i - rest * 21u;
        float v = (wp < 18u) ? xb[rest * 18u + wp] : 0.f;
        sxs[i] = (__bf16)v;
    }
    if (tid == 0) ssum = 0.f;
    __syncthreads();

    // ---- per-lane row (A-matrix row M = lane & 15) ----
    const int  row = lane & 15;
    const bool hi  = lane >= 16;
    const int  iwb = hi ? 8 : 0;

    const bf16x16* bfr = reinterpret_cast<const bf16x16*>(wsB) + lane;

    float partial = 0.f;

    for (int tile = wave; tile < NTILES; tile += 8) {
        // output position p -> (sx,sy,sz,sw), base offset in padded LDS image
        int p   = tile * 16 + row;
        int sw  = p % 6;  int t1 = p / 6;
        int sz  = t1 % 6; int t2 = t1 / 6;
        int sy  = t2 % 6; int sxp = t2 / 6;
        int base = ((sxp * 18 + sy) * 18 + sz) * 21 + sw + iwb;

        floatx8 acc = {};

        for (int t = 0; t < KSTEPS; ++t) {
            int2 c = ctab[t];               // uniform -> scalar load

            // ---- A fragment: 2x ds_load_b128 worth of bf16 gathers ----
            bf16x16 av;
            int a0 = base + c.x;
            int a1 = base + c.y;
#pragma unroll
            for (int j = 0; j < 8; ++j) av[j]     = sxs[a0 + j];
#pragma unroll
            for (int j = 0; j < 8; ++j) av[8 + j] = sxs[a1 + j];

            // ---- B fragment: 32B from L2-resident packed weights ----
            bf16x16 bv = bfr[t * 32];

            acc = __builtin_amdgcn_wmma_f32_16x16x32_bf16(
                      false, av, false, bv, (short)0, acc, false, false);
        }

        // ---- epilogue: bias + ReLU + fused linear(3888->1) partial dot ----
        int n = lane & 15;                 // D column = channel
        if (n < 3) {
            float cb = conv_b[n];
#pragma unroll
            for (int r = 0; r < 8; ++r) {
                int m  = r + (hi ? 8 : 0); // D row within tile
                int pp = tile * 16 + m;    // output position
                float v = acc[r] + cb;
                v = v > 0.f ? v : 0.f;
                partial += v * lin_w[n * 1296 + pp];
            }
        }
    }

    // ---- block reduction -> sigmoid ----
#pragma unroll
    for (int off = 16; off > 0; off >>= 1)
        partial += __shfl_down(partial, off, 32);
    if (lane == 0) atomicAdd(&ssum, partial);     // ds_add_f32
    __syncthreads();
    if (tid == 0) {
        float z = ssum + lin_b[0];
        out[b] = 1.f / (1.f + expf(-z));
    }
}

// ---------------------------------------------------------------------------
extern "C" void kernel_launch(void* const* d_in, const int* in_sizes, int n_in,
                              void* d_out, int out_size, void* d_ws, size_t ws_size,
                              hipStream_t stream) {
    const float* x      = (const float*)d_in[0];   // [512,1,18,18,18,18]
    const float* conv_w = (const float*)d_in[1];   // [3,1,13,13,13,13]
    const float* conv_b = (const float*)d_in[2];   // [3]
    const float* lin_w  = (const float*)d_in[3];   // [1,3888]
    const float* lin_b  = (const float*)d_in[4];   // [1]
    float* out = (float*)d_out;                    // [512,1]

    // workspace layout: [B-fragments: NFRAG * 32B][step table: KSTEPS * 8B]
    __bf16* wsB  = (__bf16*)d_ws;
    int2*   ctab = (int2*)((char*)d_ws + (size_t)NFRAG * 32);

    int prepThreads = NFRAG + KSTEPS;
    int prepBlocks  = (prepThreads + 255) / 256;
    conv4d_prep<<<prepBlocks, 256, 0, stream>>>(conv_w, wsB, ctab);

    conv4d_main<<<512, 256, 0, stream>>>(x, wsB, ctab, conv_b, lin_w, lin_b, out);
}
